// GatedLinearAttentionAdapter_85237920957011
// MI455X (gfx1250) — matrix operand order repaired and checked by reading, not compiled, not run on hardware
//
#include <hip/hip_runtime.h>
#include <hip/hip_bf16.h>

// Gated Linear Attention forward for MI455X (gfx1250, wave32, WMMA).
// v4: async global->LDS staging for the GEMM A tile (ASYNCcnt path) with the
// builtin's actual pointer-to-v4i parameter types; otherwise identical to v2/v3:
//   f16 pre-converted operands; WMMA GEMMs (64x128 WG tile, 32x32 wave tile,
//   transposed B staging so all fragment reads merge to ds_load_b128;
//   global_prefetch of next K tile); low-rank gate; chunkwise gated scan with
//   the recurrent state resident in LDS; fused RMSNorm/swish; output GEMM.
//
// Dims: B=2, T=1024, H=1024, NH=4, DK=512, DV=1024, dk=128, dv=256, LR=16.
// Workspace: ~37.9 MB f32 + ~16.8 MB f16 = ~54.7 MB.

typedef __attribute__((ext_vector_type(16))) _Float16 v16h;
typedef __attribute__((ext_vector_type(8)))  _Float16 v8h;
typedef __attribute__((ext_vector_type(4)))  _Float16 v4h;
typedef __attribute__((ext_vector_type(8)))  float    v8f;
typedef __attribute__((__vector_size__(4 * sizeof(int)))) int v4i;

#define HID   1024
#define ROWS  2048
#define DKC   512
#define DVC   1024
#define NHD   4
#define HKD   128   // dk per head
#define HVD   256   // dv per head
#define LRK   16
#define CHUNK 64
#define NCHUNK 16
#define VSLICE 64   // dv slice per scan workgroup
#define NSLICE 4

#if defined(__has_builtin)
#if __has_builtin(__builtin_amdgcn_global_load_async_to_lds_b128) && \
    __has_builtin(__builtin_amdgcn_s_wait_asynccnt)
#define USE_ASYNC_LDS 1
#endif
#endif

__device__ __forceinline__ v8f wmma_f16(v16h a, v16h b, v8f c) {
  return __builtin_amdgcn_wmma_f32_16x16x32_f16(false, a, false, b, (short)0, c,
                                                false, false);
}

// A-matrix 16x32 f16 fragment (ISA 7.12.2): lane L holds row m=L&15; half-wave
// grp=L>>4; VGPR j holds K = (j<4?0:16) + 8*grp + 2*(j&3).
// Contiguous f(m, k..k+7) reads merge into 2x ds_load_b128.
template <class F>
__device__ __forceinline__ v16h frag_a(F f, int lane) {
  int m = lane & 15, grp = lane >> 4;
  v16h a;
#pragma unroll
  for (int j = 0; j < 8; ++j) {
    int kb = ((j & 4) << 2) + grp * 8 + (j & 3) * 2;
    a[2 * j]     = f(m, kb);
    a[2 * j + 1] = f(m, kb + 1);
  }
  return a;
}

// B-matrix 32x16 f16 fragment: lane L holds col n=L&15; half-wave
// grp=L>>4; VGPR j holds K = (j<4?0:16) + 8*grp + 2*(j&3), +1.
template <class F>
__device__ __forceinline__ v16h frag_b(F f, int lane) {
  int n = lane & 15, grp = lane >> 4;
  v16h b;
#pragma unroll
  for (int j = 0; j < 8; ++j) {
    int kb = ((j & 4) << 2) + grp * 8 + (j & 3) * 2;
    b[2 * j]     = f(kb, n);
    b[2 * j + 1] = f(kb + 1, n);
  }
  return b;
}

// ---------------------------------------------------------------------------
// f32 -> f16 conversion (n multiple of 4)
// ---------------------------------------------------------------------------
__global__ __launch_bounds__(256) void cvt_f16(const float* __restrict__ in,
                                               _Float16* __restrict__ out, int n) {
  int idx = (blockIdx.x * 256 + threadIdx.x) * 4;
  if (idx >= n) return;
  float4 v = *(const float4*)(in + idx);
  v4h h = {(_Float16)v.x, (_Float16)v.y, (_Float16)v.z, (_Float16)v.w};
  *(v4h*)(out + idx) = h;
}

// ---------------------------------------------------------------------------
// C[M,N] = A[M,K] @ B[K,N], f16 in / f32 out.
// WG tile 64(M) x 128(N), 8 waves (2x4), wave tile 32x32 -> 4 accumulators.
// A tile staged via async global->LDS copy when available; B tile staged
// transposed so frag_b reads are contiguous (ds_load_b128).
// ---------------------------------------------------------------------------
__global__ __launch_bounds__(256) void wmma_gemm(const _Float16* __restrict__ A,
                                                 const _Float16* __restrict__ B,
                                                 float* __restrict__ C,
                                                 int M, int N, int K) {
  __shared__ _Float16 sA[64][40];    // [m][k], +8 pad (80B row, 16B aligned)
  __shared__ _Float16 sBT[128][40];  // [n][k] transposed
  const int tid = threadIdx.x, lane = tid & 31, wid = tid >> 5;
  const int bm = blockIdx.x * 64, bn = blockIdx.y * 128;
  const int wm = (wid >> 2) * 32, wn = (wid & 3) * 32;
  v8f acc[2][2] = {};
  const int ra = tid >> 2, ca = (tid & 3) * 8;    // A stage: 8 halves/thread
  const int rb = tid >> 3, cb = (tid & 7) * 16;   // B stage: 16 halves/thread
  for (int k0 = 0; k0 < K; k0 += 32) {
    __syncthreads();
    {  // stage A 64x32 (16B per lane)
      const _Float16* gA = A + (size_t)(bm + ra) * K + k0 + ca;
      _Float16* lA = &sA[ra][ca];
#ifdef USE_ASYNC_LDS
      __builtin_amdgcn_global_load_async_to_lds_b128(
          (__attribute__((address_space(1))) v4i*)(uintptr_t)gA,
          (__attribute__((address_space(3))) v4i*)(uintptr_t)lA, 0, 0);
#else
      *(v8h*)lA = *(const v8h*)gA;
#endif
    }
    {  // stage B 32x128, transposed into sBT
      v8h b0 = *(const v8h*)(B + (size_t)(k0 + rb) * N + bn + cb);
      v8h b1 = *(const v8h*)(B + (size_t)(k0 + rb) * N + bn + cb + 8);
#pragma unroll
      for (int i = 0; i < 8; ++i) sBT[cb + i][rb] = b0[i];
#pragma unroll
      for (int i = 0; i < 8; ++i) sBT[cb + 8 + i][rb] = b1[i];
    }
    if (k0 + 32 < K) {  // prefetch next K tile -> global_prefetch_b8
      __builtin_prefetch(A + (size_t)(bm + ra) * K + k0 + 32 + ca, 0, 1);
      __builtin_prefetch(B + (size_t)(k0 + 32 + rb) * N + bn + cb, 0, 1);
    }
#ifdef USE_ASYNC_LDS
    __builtin_amdgcn_s_wait_asynccnt(0);
#endif
    __syncthreads();
    v16h fa[2], fb[2];
#pragma unroll
    for (int mt = 0; mt < 2; ++mt)
      fa[mt] = frag_a([&](int m2, int kk) { return sA[wm + mt * 16 + m2][kk]; }, lane);
#pragma unroll
    for (int nt = 0; nt < 2; ++nt)
      fb[nt] = frag_b([&](int kk, int n2) { return sBT[wn + nt * 16 + n2][kk]; }, lane);
#pragma unroll
    for (int mt = 0; mt < 2; ++mt)
#pragma unroll
      for (int nt = 0; nt < 2; ++nt)
        acc[mt][nt] = wmma_f16(fa[mt], fb[nt], acc[mt][nt]);
  }
  const int grp = lane >> 4, n = lane & 15;
#pragma unroll
  for (int mt = 0; mt < 2; ++mt)
#pragma unroll
    for (int nt = 0; nt < 2; ++nt)
#pragma unroll
      for (int r = 0; r < 8; ++r) {
        int m = r + 8 * grp;
        C[(size_t)(bm + wm + mt * 16 + m) * N + bn + wn + nt * 16 + n] =
            acc[mt][nt][r];
      }
}

// ---------------------------------------------------------------------------
// Low-rank gate: t1 = x @ Wgk1   [ROWS, 16]
// ---------------------------------------------------------------------------
__global__ void gk_lr1(const float* __restrict__ x, const float* __restrict__ W1,
                       float* __restrict__ t1) {
  int row = blockIdx.x, j = threadIdx.x;
  if (j >= LRK) return;
  float acc = 0.f;
  for (int kk = 0; kk < HID; ++kk) acc += x[(size_t)row * HID + kk] * W1[kk * LRK + j];
  t1[row * LRK + j] = acc;
}

// gk = log_sigmoid(t1 @ Wgk2 + b) / 16   [ROWS, 512]
__global__ __launch_bounds__(256) void gk_lr2(const float* __restrict__ t1,
                                              const float* __restrict__ W2,
                                              const float* __restrict__ bias,
                                              float* __restrict__ gk) {
  int idx = blockIdx.x * 256 + threadIdx.x;
  int row = idx >> 9, n = idx & 511;
  float z = bias[n];
#pragma unroll
  for (int r = 0; r < LRK; ++r) z += t1[row * LRK + r] * W2[r * DKC + n];
  float ls = fminf(z, 0.f) - log1pf(__expf(-fabsf(z)));  // stable log_sigmoid
  gk[idx] = ls * (1.f / 16.f);
}

// ---------------------------------------------------------------------------
// Chunkwise gated scan. One WG per (b, head, dv-slice): 32 WGs.
// State ST[v][k] (f32, 32 KB) lives in LDS for the whole sequence.
//   G_t[k] = inclusive cumsum(gk);  qt=q*e^G*scale, kt=k*e^-G, kdT=k*e^(Gc-G)
//   A  = tril(qt @ kt^T)           o = qt @ S + A @ v
//   S  = diag(e^Gc) S + kd^T @ v
// All LDS operand layouts chosen so fragment reads are K-contiguous.
// ---------------------------------------------------------------------------
__global__ __launch_bounds__(256) void gla_scan(const float* __restrict__ q,
                                                const float* __restrict__ k,
                                                const float* __restrict__ v,
                                                const float* __restrict__ gk,
                                                float* __restrict__ o) {
  extern __shared__ __align__(16) char smem[];
  float*    ST  = (float*)smem;                // [VSLICE][HKD] f32 = 32 KB
  float*    G   = ST + VSLICE * HKD;           // [t][k] f32       = 32 KB
  float*    Gc  = G + CHUNK * HKD;             // [k] f32
  _Float16* qt  = (_Float16*)(Gc + HKD);       // [t][k]  16 KB
  _Float16* kt  = qt + CHUNK * HKD;            // [t][k]  16 KB
  _Float16* kdT = kt + CHUNK * HKD;            // [k][t]  16 KB
  _Float16* vtT = kdT + HKD * CHUNK;           // [v][t]   8 KB
  _Float16* Af  = vtT + VSLICE * CHUNK;        // [t][s]   8 KB

  const int tid = threadIdx.x, lane = tid & 31, wid = tid >> 5;
  const int vs = blockIdx.x & 3;
  const int h  = (blockIdx.x >> 2) & 3;
  const int b  = blockIdx.x >> 4;
  const int vcol0 = h * HVD + vs * VSLICE;
  const float scale = 0.08838834764831845f;  // 128^-0.5
  const int grp = lane >> 4, ln = lane & 15;

  for (int i = tid; i < VSLICE * HKD; i += 256) ST[i] = 0.f;

  for (int c = 0; c < NCHUNK; ++c) {
    const int rowbase = b * 1024 + c * CHUNK;
    __syncthreads();
    // 1) inclusive cumulative gates per k-dim
    if (tid < HKD) {
      float acc = 0.f;
      for (int t = 0; t < CHUNK; ++t) {
        acc += gk[(size_t)(rowbase + t) * DKC + h * HKD + tid];
        G[t * HKD + tid] = acc;
      }
      Gc[tid] = acc;
    }
    __syncthreads();
    // 2) build f16 operand tiles
    for (int i = tid; i < CHUNK * HKD; i += 256) {
      int t = i >> 7, kk = i & 127;
      float gcur = G[i];
      float qv = q[(size_t)(rowbase + t) * DKC + h * HKD + kk];
      float kv = k[(size_t)(rowbase + t) * DKC + h * HKD + kk];
      qt[i] = (_Float16)(qv * __expf(gcur) * scale);
      kt[i] = (_Float16)(kv * __expf(-gcur));
      kdT[kk * CHUNK + t] = (_Float16)(kv * __expf(Gc[kk] - gcur));
    }
    for (int i = tid; i < CHUNK * VSLICE; i += 256) {
      int t = i >> 6, j = i & 63;
      vtT[j * CHUNK + t] = (_Float16)v[(size_t)(rowbase + t) * DVC + vcol0 + j];
    }
    __syncthreads();
    // 3) A = tril(qt @ kt^T), 4x4 grid of 16x16 tiles
    for (int tt = wid; tt < 16; tt += 8) {
      int ti = tt >> 2, si = tt & 3;
      v8f acc = {};
      if (ti >= si) {
#pragma unroll
        for (int kb = 0; kb < HKD; kb += 32) {
          v16h fa = frag_a([&](int m2, int kk) {
            return qt[(ti * 16 + m2) * HKD + kb + kk]; }, lane);
          v16h fb = frag_b([&](int kk, int n2) {
            return kt[(si * 16 + n2) * HKD + kb + kk]; }, lane);  // kt^T
          acc = wmma_f16(fa, fb, acc);
        }
      }
#pragma unroll
      for (int r = 0; r < 8; ++r) {
        int t = ti * 16 + r + 8 * grp, s = si * 16 + ln;
        Af[t * CHUNK + s] = (_Float16)((s <= t) ? acc[r] : 0.f);
      }
    }
    __syncthreads();
    // 4) o = qt @ S + Af @ vt, 4x4 grid of 16x16 tiles
    for (int tt = wid; tt < 16; tt += 8) {
      int ti = tt >> 2, vi = tt & 3;
      v8f acc = {};
#pragma unroll
      for (int kb = 0; kb < HKD; kb += 32) {  // inter-chunk: state
        v16h fa = frag_a([&](int m2, int kk) {
          return qt[(ti * 16 + m2) * HKD + kb + kk]; }, lane);
        v16h fb = frag_b([&](int kk, int n2) {
          return (_Float16)ST[(vi * 16 + n2) * HKD + kb + kk]; }, lane);
        acc = wmma_f16(fa, fb, acc);
      }
#pragma unroll
      for (int kb = 0; kb < CHUNK; kb += 32) {  // intra-chunk
        v16h fa = frag_a([&](int m2, int kk) {
          return Af[(ti * 16 + m2) * CHUNK + kb + kk]; }, lane);
        v16h fb = frag_b([&](int kk, int n2) {
          return vtT[(vi * 16 + n2) * CHUNK + kb + kk]; }, lane);
        acc = wmma_f16(fa, fb, acc);
      }
#pragma unroll
      for (int r = 0; r < 8; ++r) {
        int t = ti * 16 + r + 8 * grp;
        o[(size_t)(rowbase + t) * DVC + vcol0 + vi * 16 + ln] = acc[r];
      }
    }
    __syncthreads();
    // 5) S = diag(exp(Gc)) S + kd^T @ vt, 8x4 grid of 16x16 tiles
    for (int tt = wid; tt < 32; tt += 8) {
      int ki = tt >> 2, vi = tt & 3;
      v8f acc;
#pragma unroll
      for (int r = 0; r < 8; ++r) {
        int kg = ki * 16 + r + 8 * grp;
        acc[r] = ST[(vi * 16 + ln) * HKD + kg] * __expf(Gc[kg]);
      }
#pragma unroll
      for (int kb = 0; kb < CHUNK; kb += 32) {
        v16h fa = frag_a([&](int m2, int t2) {
          return kdT[(ki * 16 + m2) * CHUNK + kb + t2]; }, lane);  // kd^T
        v16h fb = frag_b([&](int t2, int n2) {
          return vtT[(vi * 16 + n2) * CHUNK + kb + t2]; }, lane);
        acc = wmma_f16(fa, fb, acc);
      }
#pragma unroll
      for (int r = 0; r < 8; ++r) {
        int kg = ki * 16 + r + 8 * grp;
        ST[(vi * 16 + ln) * HKD + kg] = acc[r];
      }
    }
  }
}

// ---------------------------------------------------------------------------
// Fused RMSNorm (over dv per head) + swish gate; writes f16 for final GEMM.
// ---------------------------------------------------------------------------
__global__ __launch_bounds__(256) void rmsnorm_gate(const float* __restrict__ o,
                                                    const float* __restrict__ g,
                                                    const float* __restrict__ gw,
                                                    _Float16* __restrict__ on) {
  __shared__ float red[256];
  int row = blockIdx.x, tid = threadIdx.x;
  for (int h = 0; h < NHD; ++h) {
    float val = o[(size_t)row * DVC + h * HVD + tid];
    red[tid] = val * val;
    __syncthreads();
    for (int s2 = 128; s2 > 0; s2 >>= 1) {
      if (tid < s2) red[tid] += red[tid + s2];
      __syncthreads();
    }
    float ms = red[0] * (1.f / HVD);
    __syncthreads();
    float rstd = rsqrtf(ms + 1e-5f);
    float gv = g[(size_t)row * DVC + h * HVD + tid];
    float sig = 1.f / (1.f + __expf(-gv));
    on[(size_t)row * DVC + h * HVD + tid] =
        (_Float16)(val * rstd * gw[tid] * gv * sig);
  }
}

// ---------------------------------------------------------------------------
extern "C" void kernel_launch(void* const* d_in, const int* in_sizes, int n_in,
                              void* d_out, int out_size, void* d_ws, size_t ws_size,
                              hipStream_t stream) {
  const float* x    = (const float*)d_in[0];
  const float* Wq   = (const float*)d_in[1];
  const float* Wk   = (const float*)d_in[2];
  const float* Wv   = (const float*)d_in[3];
  const float* Wgk1 = (const float*)d_in[4];
  const float* Wgk2 = (const float*)d_in[5];
  const float* bgk2 = (const float*)d_in[6];
  const float* Wg   = (const float*)d_in[7];
  const float* Wo   = (const float*)d_in[8];
  const float* gw   = (const float*)d_in[9];

  // f32 scratch
  float* q    = (float*)d_ws;
  float* kbuf = q + (size_t)ROWS * DKC;
  float* vbuf = kbuf + (size_t)ROWS * DKC;
  float* gbuf = vbuf + (size_t)ROWS * DVC;
  float* gkb  = gbuf + (size_t)ROWS * DVC;
  float* t1   = gkb + (size_t)ROWS * DKC;
  float* obuf = t1 + (size_t)ROWS * LRK;
  // f16 scratch
  _Float16* xh  = (_Float16*)(obuf + (size_t)ROWS * DVC);
  _Float16* wqh = xh + (size_t)ROWS * HID;
  _Float16* wkh = wqh + (size_t)HID * DKC;
  _Float16* wvh = wkh + (size_t)HID * DKC;
  _Float16* wgh = wvh + (size_t)HID * DVC;
  _Float16* woh = wgh + (size_t)HID * DVC;
  _Float16* onh = woh + (size_t)DVC * HID;
  // total ws need ~54.7 MB

  // one-time f16 conversions (memory-bound GEMMs -> halve operand traffic)
  cvt_f16<<<(ROWS * HID) / 1024, 256, 0, stream>>>(x, xh, ROWS * HID);
  cvt_f16<<<(HID * DKC) / 1024, 256, 0, stream>>>(Wq, wqh, HID * DKC);
  cvt_f16<<<(HID * DKC) / 1024, 256, 0, stream>>>(Wk, wkh, HID * DKC);
  cvt_f16<<<(HID * DVC) / 1024, 256, 0, stream>>>(Wv, wvh, HID * DVC);
  cvt_f16<<<(HID * DVC) / 1024, 256, 0, stream>>>(Wg, wgh, HID * DVC);
  cvt_f16<<<(DVC * HID) / 1024, 256, 0, stream>>>(Wo, woh, DVC * HID);

  // projections (WG tile 64x128)
  wmma_gemm<<<dim3(ROWS / 64, DKC / 128), 256, 0, stream>>>(xh, wqh, q, ROWS, DKC, HID);
  wmma_gemm<<<dim3(ROWS / 64, DKC / 128), 256, 0, stream>>>(xh, wkh, kbuf, ROWS, DKC, HID);
  wmma_gemm<<<dim3(ROWS / 64, DVC / 128), 256, 0, stream>>>(xh, wvh, vbuf, ROWS, DVC, HID);
  wmma_gemm<<<dim3(ROWS / 64, DVC / 128), 256, 0, stream>>>(xh, wgh, gbuf, ROWS, DVC, HID);

  // low-rank gate
  gk_lr1<<<ROWS, 64, 0, stream>>>(x, Wgk1, t1);
  gk_lr2<<<(ROWS * DKC) / 256, 256, 0, stream>>>(t1, Wgk2, bgk2, gkb);

  // chunkwise gated scan (state resident in LDS)
  const int smem = (VSLICE * HKD + CHUNK * HKD + HKD) * 4 +
                   (2 * CHUNK * HKD + HKD * CHUNK + VSLICE * CHUNK + CHUNK * CHUNK) * 2;
  (void)hipFuncSetAttribute((const void*)gla_scan,
                            hipFuncAttributeMaxDynamicSharedMemorySize, smem);
  gla_scan<<<2 * NHD * NSLICE, 256, smem, stream>>>(q, kbuf, vbuf, gkb, obuf);

  // epilogue + output projection
  rmsnorm_gate<<<ROWS, 256, 0, stream>>>(obuf, gbuf, gw, onh);
  wmma_gemm<<<dim3(ROWS / 64, HID / 128), 256, 0, stream>>>(onh, woh, (float*)d_out,
                                                            ROWS, HID, HID);
}
